// MacNet_15530601742478
// MI455X (gfx1250) — matrix operand order repaired and checked by reading, not compiled, not run on hardware
//
#include <hip/hip_runtime.h>

// ---------------------------------------------------------------------------
// MacNet forward for MI455X (gfx1250): bf16 WMMA GEMMs + persistent GRU scan.
// GEMM staging uses CDNA5 async global->LDS loads (ASYNCcnt).
// ---------------------------------------------------------------------------

#define Hh 256
#define H2 512
#define H3 768
#define H4 1024
#define Bz 64
#define Sz 512
#define Qz 64
#define Vz 32000
#define NSTEP 8

typedef __attribute__((ext_vector_type(16))) __bf16 v16bf;
typedef __attribute__((ext_vector_type(8)))  __bf16 v8bf;
typedef __attribute__((ext_vector_type(8)))  float  v8f;

union Frag { v16bf v; v8bf h[2]; };

enum : int { GRU_LDS = 64 * 256 * 2 + 64 * 256 * 4 + 64 * 768 * 4 }; // 294912 B

// CDNA5 async copy: 16B/lane global -> LDS, tracked by ASYNCcnt.
// lds_off = low 32 bits of flat shared-aperture address (= LDS byte offset).
static __device__ __forceinline__ void async_g2l_b128(unsigned lds_off, const void* gptr) {
    asm volatile("global_load_async_to_lds_b128 %0, %1, off"
                 :: "v"(lds_off), "v"(gptr) : "memory");
}
static __device__ __forceinline__ void async_wait0() {
    asm volatile("s_wait_asynccnt 0x0" ::: "memory");
}

// ---------------------------------------------------------------------------
// f32 -> bf16 convert
__global__ void k_f2b(const float* __restrict__ x, __bf16* __restrict__ y, size_t n) {
    size_t i = (size_t)blockIdx.x * 256 + threadIdx.x;
    if (i < n) y[i] = (__bf16)x[i];
}

__global__ void k_zerof(float* __restrict__ p, size_t n) {
    size_t i = (size_t)blockIdx.x * 256 + threadIdx.x;
    if (i < n) p[i] = 0.f;
}

// embedding gather -> time-major bf16 [T,B,256]
__global__ void k_embed(const int* __restrict__ idx, const float* __restrict__ emb,
                        __bf16* __restrict__ out, int T, int Bn) {
    int i = blockIdx.x;           // t*B + b
    int t = i / Bn, b = i % Bn;
    int tok = idx[b * T + t];
    const float* src = emb + (size_t)tok * Hh;
    __bf16* dst = out + (size_t)i * Hh;
    dst[threadIdx.x] = (__bf16)src[threadIdx.x];
}

// ---------------------------------------------------------------------------
// Generic GEMM: C[M,N] = act(A_bf16[M,K] @ W_bf16[N,K]^T + bias).
// 64x128 block tile, 8 waves x (1 m-tile x 4 n-tiles), K step 32.
// Staging via async global->LDS (no VGPR round-trip); A-frag reused 4x.
__global__ __launch_bounds__(256) void k_gemm(
    const __bf16* __restrict__ A, int lda,
    const __bf16* __restrict__ W, int ldw,
    const float* __restrict__ bias,
    float* __restrict__ Cf, __bf16* __restrict__ Cb, int ldc,
    int K, int act)
{
    __shared__ __bf16 As[64][32];    // 4 KB
    __shared__ __bf16 Ws[128][32];   // 8 KB
    const int tid = threadIdx.x, lane = tid & 31, wid = tid >> 5;
    const int bm = blockIdx.y << 6, bn = blockIdx.x << 7;
    const int wm = wid & 3, wg = wid >> 2;         // 4 m-tiles x 2 n-groups(x4)
    const int r16 = lane & 15, hi = lane >> 4;
    const int lr = tid >> 2, lc = (tid & 3) << 3;  // cooperative tile load
    const unsigned lds_a  = (unsigned)(size_t)&As[lr][lc];
    const unsigned lds_w0 = (unsigned)(size_t)&Ws[lr][lc];
    const unsigned lds_w1 = (unsigned)(size_t)&Ws[lr + 64][lc];
    v8f acc[4];
    const v8f vzero = {0.f,0.f,0.f,0.f,0.f,0.f,0.f,0.f};
    for (int j = 0; j < 4; ++j) acc[j] = vzero;
    const int nk = K >> 5;
    for (int kk = 0; kk < nk; ++kk) {
        const __bf16* ga  = &A[(size_t)(bm + lr) * lda + (kk << 5) + lc];
        const __bf16* gw0 = &W[(size_t)(bn + lr) * ldw + (kk << 5) + lc];
        const __bf16* gw1 = &W[(size_t)(bn + 64 + lr) * ldw + (kk << 5) + lc];
        async_g2l_b128(lds_a, ga);
        async_g2l_b128(lds_w0, gw0);
        async_g2l_b128(lds_w1, gw1);
        if (kk + 1 < nk) {
            __builtin_prefetch(&A[(size_t)(bm + lr) * lda + ((kk + 1) << 5) + lc], 0, 3);
            __builtin_prefetch(&W[(size_t)(bn + lr) * ldw + ((kk + 1) << 5) + lc], 0, 3);
        }
        async_wait0();
        __syncthreads();
        Frag fa;
        const __bf16* ar = &As[wm * 16 + r16][hi * 8];       // A: lanes<16 K{0-7,16-23}
        fa.h[0] = *(const v8bf*)ar;
        fa.h[1] = *(const v8bf*)(ar + 16);
        for (int j = 0; j < 4; ++j) {
            Frag fb;                                          // B: lane&15=N, half by lane>=16
            const __bf16* br = &Ws[(wg * 4 + j) * 16 + r16][hi * 8];
            fb.h[0] = *(const v8bf*)br;
            fb.h[1] = *(const v8bf*)(br + 16);
            acc[j] = __builtin_amdgcn_wmma_f32_16x16x32_bf16(false, fa.v, false, fb.v,
                                                             (short)0, acc[j], false, false);
        }
        __syncthreads();
    }
    const int m0 = bm + wm * 16 + hi * 8;
    for (int j = 0; j < 4; ++j) {
        int n = bn + (wg * 4 + j) * 16 + r16;
        float bv = bias ? bias[n] : 0.f;
        for (int v = 0; v < 8; ++v) {
            float x = acc[j][v] + bv;
            if (act == 1) x = fmaxf(x, 0.f);
            size_t cix = (size_t)(m0 + v) * ldc + n;
            if (Cf) Cf[cix] = x;
            if (Cb) Cb[cix] = (__bf16)x;
        }
    }
}

// ---------------------------------------------------------------------------
// Persistent bidirectional GRU scan. grid=2 (blockIdx.x: 0=fwd, 1=bwd),
// 1024 threads = 32 waves. Per step: gh[64,768] = h_bf16 @ Whh^T via 192 WMMA
// tiles, then fused gates. h (f32+bf16) and gh live in 288KB dynamic LDS.
__global__ __launch_bounds__(1024) void k_gru_scan(
    const __bf16* __restrict__ gi_f, const __bf16* __restrict__ gi_b,
    const __bf16* __restrict__ whh_f, const __bf16* __restrict__ whh_b,
    const float* __restrict__ bhh_f, const float* __restrict__ bhh_b,
    float* __restrict__ out, int T)
{
    const int dir = blockIdx.x;
    const __bf16* gi  = dir ? gi_b  : gi_f;
    const __bf16* whh = dir ? whh_b : whh_f;
    const float*  bhh = dir ? bhh_b : bhh_f;
    const int off = dir ? Hh : 0;
    extern __shared__ char smem[];
    __bf16* hb = (__bf16*)smem;                                // [64][256] bf16
    float*  hf = (float*)(smem + 64 * 256 * 2);                // [64][256] f32
    float*  gh = (float*)(smem + 64 * 256 * 2 + 64 * 256 * 4); // [64][768] f32
    const int tid = threadIdx.x, lane = tid & 31, wid = tid >> 5;
    const int wm = wid & 3, wg = wid >> 2;                     // 4 m-tiles x 8 n-groups(x6)
    const int r16 = lane & 15, hi = lane >> 4;
    const v8f vzero = {0.f,0.f,0.f,0.f,0.f,0.f,0.f,0.f};

    for (int e = tid; e < 64 * 256; e += 1024) { hf[e] = 0.f; hb[e] = (__bf16)0.f; }
    __syncthreads();

    for (int s = 0; s < T; ++s) {
        const int t = dir ? (T - 1 - s) : s;
        v8f acc[6];
        for (int j = 0; j < 6; ++j) acc[j] = vzero;
        for (int kk = 0; kk < 8; ++kk) {                       // K = 256
            Frag fa;
            const __bf16* ar = hb + (wm * 16 + r16) * 256 + kk * 32;
            fa.h[0] = *(const v8bf*)(ar + hi * 8);
            fa.h[1] = *(const v8bf*)(ar + hi * 8 + 16);
            for (int j = 0; j < 6; ++j) {
                const int n0 = (wg * 6 + j) * 16;
                Frag fb;
                const __bf16* br = whh + (size_t)(n0 + r16) * 256 + kk * 32 + hi * 8;
                fb.h[0] = *(const v8bf*)br;
                fb.h[1] = *(const v8bf*)(br + 16);
                acc[j] = __builtin_amdgcn_wmma_f32_16x16x32_bf16(false, fa.v, false, fb.v,
                                                                 (short)0, acc[j], false, false);
            }
        }
        for (int j = 0; j < 6; ++j) {
            const int n0 = (wg * 6 + j) * 16;
            const int m = wm * 16 + hi * 8;
            for (int v = 0; v < 8; ++v)
                gh[(m + v) * 768 + n0 + r16] = acc[j][v];
        }
        __syncthreads();
        const __bf16* git = gi + (size_t)t * 64 * 768;
        for (int e = tid; e < 64 * 256; e += 1024) {
            const int b = e >> 8, d = e & 255;
            float gir = (float)git[b * 768 + d];
            float giz = (float)git[b * 768 + 256 + d];
            float gin = (float)git[b * 768 + 512 + d];
            float ghr = gh[b * 768 + d]       + bhh[d];
            float ghz = gh[b * 768 + 256 + d] + bhh[256 + d];
            float ghn = gh[b * 768 + 512 + d] + bhh[512 + d];
            float r = 1.f / (1.f + __expf(-(gir + ghr)));
            float z = 1.f / (1.f + __expf(-(giz + ghz)));
            float n = tanhf(gin + r * ghn);
            float hn = (1.f - z) * n + z * hf[e];
            hf[e] = hn; hb[e] = (__bf16)hn;
            out[((size_t)t * 64 + b) * H2 + off + d] = hn;
        }
        __syncthreads();
    }
}

// ---------------------------------------------------------------------------
// out[i] = bias + sum_d w[d] * (g?g[b,d]:1) * x[i,d]     (i = t*B + b)
__global__ void k_attend(const float* __restrict__ g, const float* __restrict__ x,
                         const float* __restrict__ w, const float* __restrict__ bias,
                         float* __restrict__ out, int Bn, int D) {
    int i = blockIdx.x;
    int b = i % Bn;
    __shared__ float red[128];
    float s = 0.f;
    for (int d = threadIdx.x; d < D; d += 128) {
        float gv = g ? g[(size_t)b * D + d] : 1.f;
        s += w[d] * gv * x[(size_t)i * D + d];
    }
    red[threadIdx.x] = s; __syncthreads();
    for (int st = 64; st > 0; st >>= 1) {
        if (threadIdx.x < st) red[threadIdx.x] += red[threadIdx.x + st];
        __syncthreads();
    }
    if (threadIdx.x == 0) out[i] = red[0] + bias[0];
}

// softmax over axis0 of [T,B]
__global__ void k_softmax_col(const float* __restrict__ lgt, float* __restrict__ out,
                              int T, int Bn) {
    int b = blockIdx.x;
    __shared__ float red[256];
    float mx = -1e30f;
    for (int t = threadIdx.x; t < T; t += 256) mx = fmaxf(mx, lgt[(size_t)t * Bn + b]);
    red[threadIdx.x] = mx; __syncthreads();
    for (int st = 128; st > 0; st >>= 1) {
        if (threadIdx.x < st) red[threadIdx.x] = fmaxf(red[threadIdx.x], red[threadIdx.x + st]);
        __syncthreads();
    }
    mx = red[0]; __syncthreads();
    float sm = 0.f;
    for (int t = threadIdx.x; t < T; t += 256) sm += __expf(lgt[(size_t)t * Bn + b] - mx);
    red[threadIdx.x] = sm; __syncthreads();
    for (int st = 128; st > 0; st >>= 1) {
        if (threadIdx.x < st) red[threadIdx.x] += red[threadIdx.x + st];
        __syncthreads();
    }
    float inv = 1.f / red[0];
    for (int t = threadIdx.x; t < T; t += 256)
        out[(size_t)t * Bn + b] = __expf(lgt[(size_t)t * Bn + b] - mx) * inv;
}

// out[b,d] = sum_t w[t,b] * x[t,b,d]
__global__ void k_wsum(const float* __restrict__ w, const float* __restrict__ x,
                       float* __restrict__ out, int T, int Bn, int D) {
    int b = blockIdx.x;
    for (int d = threadIdx.x; d < D; d += 256) {
        float acc = 0.f;
        for (int t = 0; t < T; ++t)
            acc += w[(size_t)t * Bn + b] * x[((size_t)t * Bn + b) * D + d];
        out[(size_t)b * D + d] = acc;
    }
}

// out[i,0:512] = bf16(pm[b]*ps[i]); out[i,512:1024] = bf16(story[i])
__global__ void k_mulcat(const float* __restrict__ pm, const float* __restrict__ ps,
                         const float* __restrict__ st, __bf16* __restrict__ out, int Bn) {
    size_t i = blockIdx.x;
    int b = (int)(i % Bn);
    __bf16* o = out + i * (size_t)H4;
    const float* pr = ps + i * (size_t)H2;
    const float* sr = st + i * (size_t)H2;
    const float* pmr = pm + (size_t)b * H2;
    for (int d = threadIdx.x; d < H2; d += 256) {
        o[d]      = (__bf16)(pmr[d] * pr[d]);
        o[H2 + d] = (__bf16)sr[d];
    }
}

__global__ void k_concat2(const float* __restrict__ x, const float* __restrict__ y,
                          __bf16* __restrict__ out, int D1, int D2) {
    int b = blockIdx.x;
    __bf16* o = out + (size_t)b * (D1 + D2);
    for (int d = threadIdx.x; d < D1; d += 256) o[d] = (__bf16)x[(size_t)b * D1 + d];
    for (int d = threadIdx.x; d < D2; d += 256)
        o[D1 + d] = (__bf16)(y ? y[(size_t)b * D2 + d] : 0.f);
}

__global__ void k_concat3(const float* __restrict__ x, const float* __restrict__ y,
                          const float* __restrict__ z, __bf16* __restrict__ out, int D) {
    int b = blockIdx.x;
    __bf16* o = out + (size_t)b * 3 * D;
    for (int d = threadIdx.x; d < D; d += 256) {
        o[d]         = (__bf16)x[(size_t)b * D + d];
        o[D + d]     = (__bf16)y[(size_t)b * D + d];
        o[2 * D + d] = (__bf16)z[(size_t)b * D + d];
    }
}

// mi = pc*m_prev + (1-pc)*(mipA+mipB); mattn[b] = pc
__global__ void k_write_mi(const float* __restrict__ pcl, const float* __restrict__ mprev,
                           const float* __restrict__ a, const float* __restrict__ b2,
                           float* __restrict__ mnext, float* __restrict__ matt) {
    int e = blockIdx.x * 256 + threadIdx.x;     // < 64*512
    int bb = e >> 9, d = e & 511;
    float pc = 1.f / (1.f + __expf(-pcl[bb]));
    mnext[e] = pc * mprev[e] + (1.f - pc) * (a[e] + b2[e]);
    if (d == 0) matt[bb] = pc;
}

// log_softmax over V per row
__global__ void k_logsm(const float* __restrict__ x, float* __restrict__ out, int V) {
    int b = blockIdx.x;
    const float* xb = x + (size_t)b * V;
    __shared__ float red[256];
    float mx = -1e30f;
    for (int v = threadIdx.x; v < V; v += 256) mx = fmaxf(mx, xb[v]);
    red[threadIdx.x] = mx; __syncthreads();
    for (int st = 128; st > 0; st >>= 1) {
        if (threadIdx.x < st) red[threadIdx.x] = fmaxf(red[threadIdx.x], red[threadIdx.x + st]);
        __syncthreads();
    }
    mx = red[0]; __syncthreads();
    float s = 0.f;
    for (int v = threadIdx.x; v < V; v += 256) s += __expf(xb[v] - mx);
    red[threadIdx.x] = s; __syncthreads();
    for (int st = 128; st > 0; st >>= 1) {
        if (threadIdx.x < st) red[threadIdx.x] += red[threadIdx.x + st];
        __syncthreads();
    }
    float lse = mx + __logf(red[0]);
    for (int v = threadIdx.x; v < V; v += 256) out[(size_t)b * V + v] = xb[v] - lse;
}

// ---------------------------------------------------------------------------
struct Dir { __bf16 *wih, *whh; const float *bih, *bhh; };

extern "C" void kernel_launch(void* const* d_in, const int* in_sizes, int n_in,
                              void* d_out, int out_size, void* d_ws, size_t ws_size,
                              hipStream_t stream) {
    (void)in_sizes; (void)n_in; (void)out_size; (void)ws_size;
    // d_in flattened in setup_inputs() insertion order:
    // 0 story, 1 question, 2 embed, 3..18 gru_story (L0 fwd/bwd, L1 fwd/bwd:
    // Wih,Whh,bih,bhh), 19..34 gru_question, then linears (W,b pairs) 35..62.
    const int*   story    = (const int*)d_in[0];
    const int*   question = (const int*)d_in[1];
    const float* embed    = (const float*)d_in[2];
    auto F = [&](int i) -> const float* { return (const float*)d_in[i]; };

    size_t off = 0;
    auto alloc = [&](size_t bytes) -> void* {
        size_t a = (off + 255) & ~(size_t)255;
        off = a + bytes;
        return (void*)((char*)d_ws + a);
    };
    auto cvt = [&](const float* src, size_t n) -> __bf16* {
        __bf16* p = (__bf16*)alloc(n * sizeof(__bf16));
        k_f2b<<<dim3((unsigned)((n + 255) / 256)), 256, 0, stream>>>(src, p, n);
        return p;
    };
    auto cvt_in = [&](int i, size_t n) -> __bf16* { return cvt(F(i), n); };
    auto gemm = [&](const __bf16* A, int lda, const __bf16* W, int ldw, const float* bias,
                    float* Cf, __bf16* Cb, int ldc, int M, int N, int K, int act) {
        k_gemm<<<dim3((unsigned)(N / 128), (unsigned)(M / 64)), 256, 0, stream>>>(
            A, lda, W, ldw, bias, Cf, Cb, ldc, K, act);
    };

    // ---- weights -> bf16 (all L2-resident: ~32MB) ----
    auto mkd = [&](int base, int inK) -> Dir {
        Dir d;
        d.wih = cvt_in(base + 0, (size_t)H3 * inK);
        d.whh = cvt_in(base + 1, (size_t)H3 * Hh);
        d.bih = F(base + 2);
        d.bhh = F(base + 3);
        return d;
    };
    Dir s0f = mkd(3, Hh),  s0b = mkd(7, Hh),  s1f = mkd(11, H2), s1b = mkd(15, H2);
    Dir q0f = mkd(19, Hh), q0b = mkd(23, Hh), q1f = mkd(27, H2), q1b = mkd(31, H2);
    __bf16* w_pq    = cvt_in(35, (size_t)H2 * H4);   const float* b_pq   = F(36);
    __bf16* w_cqc   = cvt_in(37, (size_t)H2 * 1536); const float* b_cqc  = F(38);
    const float* W_ca  = F(39); const float* b_ca  = F(40);
    __bf16* w_rps   = cvt_in(41, (size_t)H2 * H2);   const float* b_rps  = F(42);
    __bf16* w_rpm   = cvt_in(43, (size_t)H2 * H2);   const float* b_rpm  = F(44);
    __bf16* w_blend = cvt_in(45, (size_t)H2 * H4);   const float* b_blend = F(46);
    const float* W_ric = F(47); const float* b_ric = F(48);
    __bf16* w_wmi   = cvt_in(49, (size_t)H2 * H4);   const float* b_wmi  = F(50);
    const float* W_wa  = F(51); const float* b_wa  = F(52);
    __bf16* w_wpm   = cvt_in(53, (size_t)H2 * H2);   const float* b_wpm  = F(54);
    __bf16* w_wpms  = cvt_in(55, (size_t)H2 * H2);   const float* b_wpms = F(56);
    const float* W_wpc = F(57); const float* b_wpc = F(58);
    __bf16* w_proj  = cvt_in(59, (size_t)Hh * H4);   const float* b_proj = F(60);
    __bf16* w_ans   = cvt_in(61, (size_t)Vz * Hh);   const float* b_ans  = F(62);

    // ---- activations / scratch ----
    __bf16* s_emb  = (__bf16*)alloc((size_t)Sz * Bz * Hh * 2);
    __bf16* q_emb  = (__bf16*)alloc((size_t)Qz * Bz * Hh * 2);
    __bf16* gi_fw  = (__bf16*)alloc((size_t)Sz * Bz * H3 * 2);  // contiguous with gi_bw:
    __bf16* gi_bw  = (__bf16*)alloc((size_t)Sz * Bz * H3 * 2);  // reused as pmps (67MB<=100MB)
    float*  x0_f   = (float*)alloc((size_t)Sz * Bz * H2 * 4);   // story L0 out, reused as Ip
    __bf16* x0_b   = (__bf16*)alloc((size_t)Sz * Bz * H2 * 2);
    float*  senc   = (float*)alloc((size_t)Sz * Bz * H2 * 4);
    __bf16* senc_b = (__bf16*)alloc((size_t)Sz * Bz * H2 * 2);
    float*  q0_f   = (float*)alloc((size_t)Qz * Bz * H2 * 4);
    __bf16* q0_b   = (__bf16*)alloc((size_t)Qz * Bz * H2 * 2);
    float*  qenc   = (float*)alloc((size_t)Qz * Bz * H2 * 4);
    float*  ps     = (float*)alloc((size_t)Sz * Bz * H2 * 4);
    float*  lgt    = (float*)alloc((size_t)Sz * Bz * 4);
    float*  saij   = (float*)alloc((size_t)(NSTEP + 1) * Bz * 4);
    float*  c_hist = (float*)alloc((size_t)(NSTEP + 1) * Bz * H2 * 4);
    float*  m_hist = (float*)alloc((size_t)(NSTEP + 1) * Bz * H2 * 4);
    float*  qiA    = (float*)alloc((size_t)Bz * H2 * 4);
    float*  qiB    = (float*)alloc((size_t)Bz * H2 * 4);
    float*  cqi    = (float*)alloc((size_t)Bz * H2 * 4);
    float*  pm     = (float*)alloc((size_t)Bz * H2 * 4);
    float*  ri     = (float*)alloc((size_t)Bz * H2 * 4);
    float*  minfo  = (float*)alloc((size_t)Bz * H2 * 4);
    float*  misa   = (float*)alloc((size_t)Bz * H2 * 4);
    float*  mipA   = (float*)alloc((size_t)Bz * H2 * 4);
    float*  mipB   = (float*)alloc((size_t)Bz * H2 * 4);
    float*  pcl    = (float*)alloc((size_t)Bz * 4);
    __bf16* m_b     = (__bf16*)alloc((size_t)Bz * H2 * 2);
    __bf16* minfo_b = (__bf16*)alloc((size_t)Bz * H2 * 2);
    __bf16* misa_b  = (__bf16*)alloc((size_t)Bz * H2 * 2);
    __bf16* cat1024 = (__bf16*)alloc((size_t)Bz * H4 * 2);
    __bf16* cat1536 = (__bf16*)alloc((size_t)Bz * 1536 * 2);
    float*  outh    = (float*)alloc((size_t)Bz * Hh * 4);
    __bf16* outh_b  = (__bf16*)alloc((size_t)Bz * Hh * 2);
    float*  ans     = (float*)alloc((size_t)Bz * Vz * 4);

    float* out_logp = (float*)d_out;
    float* out_satt = out_logp + (size_t)Bz * Vz;
    float* out_qatt = out_satt + (size_t)NSTEP * Sz * Bz;
    float* out_matt = out_qatt + (size_t)NSTEP * Qz * Bz;

    auto f2b = [&](const float* x, __bf16* y, size_t n) {
        k_f2b<<<dim3((unsigned)((n + 255) / 256)), 256, 0, stream>>>(x, y, n);
    };

    // ---- encoders ----
    k_embed<<<Sz * Bz, 256, 0, stream>>>(story, embed, s_emb, Sz, Bz);
    k_embed<<<Qz * Bz, 256, 0, stream>>>(question, embed, q_emb, Qz, Bz);

    auto encode_layer = [&](const __bf16* xin, int K, const Dir& fd, const Dir& bd, int T,
                            float* out_f, __bf16* out_b) {
        gemm(xin, K, fd.wih, K, fd.bih, nullptr, gi_fw, H3, T * Bz, H3, K, 0);
        gemm(xin, K, bd.wih, K, bd.bih, nullptr, gi_bw, H3, T * Bz, H3, K, 0);
        (void)hipFuncSetAttribute((const void*)k_gru_scan,
                                  hipFuncAttributeMaxDynamicSharedMemorySize, GRU_LDS);
        k_gru_scan<<<2, 1024, GRU_LDS, stream>>>(gi_fw, gi_bw, fd.whh, bd.whh,
                                                 fd.bhh, bd.bhh, out_f, T);
        if (out_b) f2b(out_f, out_b, (size_t)T * Bz * H2);
    };
    encode_layer(s_emb, Hh, s0f, s0b, Sz, x0_f, x0_b);
    encode_layer(x0_b,  H2, s1f, s1b, Sz, senc, senc_b);
    encode_layer(q_emb, Hh, q0f, q0b, Qz, q0_f, q0_b);
    encode_layer(q0_b,  H2, q1f, q1b, Qz, qenc, nullptr);

    // step-invariant: ps = read_project_story(story)
    gemm(senc_b, H2, w_rps, H2, b_rps, ps, nullptr, H2, Sz * Bz, H2, H2, 0);

    // c[0] = m[0] = 0; qi = produce_qi([question[-1], m0])
    k_zerof<<<(Bz * H2 + 255) / 256, 256, 0, stream>>>(c_hist, (size_t)Bz * H2);
    k_zerof<<<(Bz * H2 + 255) / 256, 256, 0, stream>>>(m_hist, (size_t)Bz * H2);
    k_concat2<<<Bz, 256, 0, stream>>>(qenc + (size_t)(Qz - 1) * Bz * H2, m_hist,
                                      cat1024, H2, H2);
    gemm(cat1024, H4, w_pq, H4, b_pq, qiA, nullptr, H2, Bz, H2, H4, 0);

    __bf16* pmps = gi_fw;   // gi_fw+gi_bw region reused for [S,B,1024] bf16
    float*  Ip   = x0_f;    // story L0 buffer reused
    float* qi = qiA;
    float* qin = qiB;

    for (int i = 0; i < NSTEP; ++i) {
        float* m_prev = m_hist + (size_t)i * Bz * H2;
        float* c_prev = c_hist + (size_t)i * Bz * H2;
        float* ci     = c_hist + (size_t)(i + 1) * Bz * H2;
        float* m_next = m_hist + (size_t)(i + 1) * Bz * H2;
        // control
        k_concat3<<<Bz, 256, 0, stream>>>(c_prev, qi, m_prev, cat1536, H2);
        gemm(cat1536, 1536, w_cqc, 1536, b_cqc, cqi, nullptr, H2, Bz, H2, 1536, 0);
        k_attend<<<Qz * Bz, 128, 0, stream>>>(cqi, qenc, W_ca, b_ca, lgt, Bz, H2);
        float* qw = out_qatt + (size_t)i * Qz * Bz;
        k_softmax_col<<<Bz, 256, 0, stream>>>(lgt, qw, Qz, Bz);
        k_wsum<<<Bz, 256, 0, stream>>>(qw, qenc, ci, Qz, Bz, H2);
        // read
        f2b(m_prev, m_b, (size_t)Bz * H2);
        gemm(m_b, H2, w_rpm, H2, b_rpm, pm, nullptr, H2, Bz, H2, H2, 0);
        k_mulcat<<<Sz * Bz, 256, 0, stream>>>(pm, ps, senc, pmps, Bz);
        gemm(pmps, H4, w_blend, H4, b_blend, Ip, nullptr, H2, Sz * Bz, H2, H4, 0);
        k_attend<<<Sz * Bz, 128, 0, stream>>>(ci, Ip, W_ric, b_ric, lgt, Bz, H2);
        float* sw = out_satt + (size_t)i * Sz * Bz;
        k_softmax_col<<<Bz, 256, 0, stream>>>(lgt, sw, Sz, Bz);
        k_wsum<<<Bz, 256, 0, stream>>>(sw, senc, ri, Sz, Bz, H2);
        // write
        k_concat2<<<Bz, 256, 0, stream>>>(m_prev, ri, cat1024, H2, H2);
        gemm(cat1024, H4, w_wmi, H4, b_wmi, minfo, nullptr, H2, Bz, H2, H4, 0);
        f2b(minfo, minfo_b, (size_t)Bz * H2);
        gemm(minfo_b, H2, w_wpm, H2, b_wpm, mipA, nullptr, H2, Bz, H2, H2, 0);
        k_attend<<<(i + 1) * Bz, 128, 0, stream>>>(ci, c_hist, W_wa, b_wa, lgt, Bz, H2);
        k_softmax_col<<<Bz, 256, 0, stream>>>(lgt, saij, i + 1, Bz);
        k_wsum<<<Bz, 256, 0, stream>>>(saij, m_hist, misa, i + 1, Bz, H2);
        f2b(misa, misa_b, (size_t)Bz * H2);
        gemm(misa_b, H2, w_wpms, H2, b_wpms, mipB, nullptr, H2, Bz, H2, H2, 0);
        k_attend<<<Bz, 128, 0, stream>>>(nullptr, ci, W_wpc, b_wpc, pcl, Bz, H2);
        k_write_mi<<<(Bz * H2) / 256, 256, 0, stream>>>(pcl, m_prev, mipA, mipB, m_next,
                                                        out_matt + (size_t)i * Bz);
        // qi update (uses OLD memory)
        k_concat2<<<Bz, 256, 0, stream>>>(qi, m_prev, cat1024, H2, H2);
        gemm(cat1024, H4, w_pq, H4, b_pq, qin, nullptr, H2, Bz, H2, H4, 0);
        float* tswap = qi; qi = qin; qin = tswap;
    }

    // output head
    k_concat2<<<Bz, 256, 0, stream>>>(qi, m_hist + (size_t)NSTEP * Bz * H2, cat1024, H2, H2);
    gemm(cat1024, H4, w_proj, H4, b_proj, outh, nullptr, Hh, Bz, Hh, H4, 1 /*relu*/);
    f2b(outh, outh_b, (size_t)Bz * Hh);
    gemm(outh_b, Hh, w_ans, Hh, b_ans, ans, nullptr, Vz, Bz, Vz, Hh, 0);
    k_logsm<<<Bz, 256, 0, stream>>>(ans, out_logp, Vz);
}
